// MultiheadDiffAttn_58334245814890
// MI455X (gfx1250) — matrix operand order repaired and checked by reading, not compiled, not run on hardware
//
#include <hip/hip_runtime.h>

// ---------------- problem constants ----------------
#define PB 2
#define PN 2048
#define PE 1024
#define PH 8
// head dim D=64, v head dim 2D=128

typedef __attribute__((ext_vector_type(16))) __bf16 v16bf;
typedef __attribute__((ext_vector_type(8)))  float  v8f;
typedef unsigned int u32x4 __attribute__((ext_vector_type(4)));
typedef int i32x4 __attribute__((ext_vector_type(4)));
typedef int i32x8 __attribute__((ext_vector_type(8)));

#if __has_builtin(__builtin_amdgcn_tensor_load_to_lds)
#define HAVE_TDM 1
#else
#define HAVE_TDM 0
#endif

union BF16Frag { v16bf v; unsigned short u[16]; unsigned int w[8]; };

__device__ __forceinline__ unsigned short f2bf(float f) {
  unsigned int u = __builtin_bit_cast(unsigned int, f);
  return (unsigned short)((u + 0x7FFFu + ((u >> 16) & 1u)) >> 16);
}

__device__ __forceinline__ v8f wmma_bf16(v16bf a, v16bf b, v8f c) {
  // v_wmma_f32_16x16x32_bf16  D = A(16x32) * B(32x16) + C
  return __builtin_amdgcn_wmma_f32_16x16x32_bf16(false, a, false, b, (short)0, c,
                                                 false, false);
}

// ---------------- TDM helpers (gfx1250 Tensor Data Mover) ----------------
// Low 32 bits of a flat shared pointer == LDS byte address (ISA 10.2 aperture).
__device__ __forceinline__ unsigned lds_off(const void* p) {
  return (unsigned)(uintptr_t)p;
}

__device__ __forceinline__ void wait_tensorcnt0() {
#if __has_builtin(__builtin_amdgcn_s_wait_tensorcnt)
  __builtin_amdgcn_s_wait_tensorcnt(0);
#else
  asm volatile("s_wait_tensorcnt 0x0" ::: "memory");
#endif
}

#if HAVE_TDM
// 2D tile load: elements are 2 bytes (bf16). dims/strides/tiles in elements.
// Tile lands contiguously in LDS: [tile1][tile0] row-major.
__device__ __forceinline__ void tdm_load_2d(unsigned lds,
                                            const unsigned short* gptr,
                                            unsigned dim0, unsigned dim1,
                                            unsigned stride0, unsigned tile0,
                                            unsigned tile1) {
  unsigned long long ga = (unsigned long long)(uintptr_t)gptr;
  u32x4 g0;
  g0[0] = 1u;                                  // count=1, user descriptor
  g0[1] = lds;                                 // lds_addr [63:32]
  g0[2] = (unsigned)ga;                        // global_addr [95:64]
  g0[3] = (unsigned)((ga >> 32) & 0x01FFFFFFull) | (2u << 30);  // type=2
  i32x8 g1;
  g1[0] = (int)(1u << 16);                     // data_size=2B; mask=0; no pad
  g1[1] = (int)((dim0 & 0xFFFFu) << 16);       // tensor_dim0[15:0] @ [63:48]
  g1[2] = (int)(((dim0 >> 16) & 0xFFFFu) | ((dim1 & 0xFFFFu) << 16));
  g1[3] = (int)(((dim1 >> 16) & 0xFFFFu) | ((tile0 & 0xFFFFu) << 16));
  g1[4] = (int)(tile1 & 0xFFFFu);              // tile_dim1; tile_dim2=0
  g1[5] = (int)stride0;                        // tensor_dim0_stride[31:0]
  g1[6] = 0;                                   // stride0[47:32], dim1_stride lo
  g1[7] = 0;                                   // dim1_stride hi
  i32x4 z4 = {0, 0, 0, 0};
#if defined(__clang_major__) && __clang_major__ >= 23
  i32x8 z8 = {0, 0, 0, 0, 0, 0, 0, 0};
  __builtin_amdgcn_tensor_load_to_lds(g0, g1, z4, z4, z8, 0);
#else
  __builtin_amdgcn_tensor_load_to_lds(g0, g1, z4, z4, 0);
#endif
}
#endif

// A fragment (16x32 bf16, MxK) from row-major [16 x ld] bf16 buffer.
// Lane L: row M = L%16, half = L/16; pair p: K = (p<4 ? 2p : 2p+8) + 8*half.
__device__ __forceinline__ v16bf load_a_frag(const unsigned short* base, int ld) {
  int lane = (int)(threadIdx.x & 31u);
  int row = lane & 15, half = lane >> 4;
  const unsigned short* p = base + row * ld + 8 * half;
  BF16Frag f;
#pragma unroll
  for (int pr = 0; pr < 8; ++pr) {
    int k = (pr < 4) ? (2 * pr) : (2 * pr + 8);
    f.w[pr] = *(const unsigned int*)(p + k);
  }
  return f.v;
}

// B fragment (KxN = 32x16): B[k][n] taken from an n-major buffer src[n][k]
// (row-major [>=16 rows(n)][ld cols(k)]).  Lane L: n = L%16, pair p: K = (p<4 ? 2p : 2p+8) + 8*half.
__device__ __forceinline__ v16bf load_b_frag(const unsigned short* src, int ld,
                                             int kofs) {
  int lane = (int)(threadIdx.x & 31u);
  int n = lane & 15, half = lane >> 4;
  const unsigned short* p = src + n * ld + kofs + 8 * half;
  BF16Frag f;
#pragma unroll
  for (int j = 0; j < 8; ++j) f.w[j] = *(const unsigned int*)(p + ((j < 4) ? (2 * j) : (2 * j + 8)));
  return f.v;
}

__device__ __forceinline__ float rmax16(float v) {
#pragma unroll
  for (int m = 1; m <= 8; m <<= 1) v = fmaxf(v, __shfl_xor(v, m, 32));
  return v;
}
__device__ __forceinline__ float rsum16(float v) {
#pragma unroll
  for (int m = 1; m <= 8; m <<= 1) v += __shfl_xor(v, m, 32);
  return v;
}

// ---------------- fp32 -> bf16 convert ----------------
__global__ void cvt_bf16_kernel(const float* __restrict__ in,
                                unsigned short* __restrict__ out, int n) {
  int i = blockIdx.x * blockDim.x + threadIdx.x;
  if (i < n) out[i] = f2bf(in[i]);
}

// fused convert + transpose for 1024x1024 weights: outT[n][k] = in[k][n]
__global__ void cvt_bf16_T_kernel(const float* __restrict__ in,
                                  unsigned short* __restrict__ outT) {
  int i = blockIdx.x * blockDim.x + threadIdx.x;  // index over outT (coalesced)
  int n = i >> 10, k = i & 1023;
  outT[i] = f2bf(in[k * 1024 + n]);
}

// ---------------- lambda scalar ----------------
__global__ void lambda_kernel(const float* lq1, const float* lk1,
                              const float* lq2, const float* lk2,
                              float* lam) {
  int lane = (int)threadIdx.x;  // 32 threads, D=64 -> 2 per lane
  float a = lq1[lane] * lk1[lane] + lq1[lane + 32] * lk1[lane + 32];
  float c = lq2[lane] * lk2[lane] + lq2[lane + 32] * lk2[lane + 32];
#pragma unroll
  for (int m = 1; m <= 16; m <<= 1) {
    a += __shfl_xor(a, m, 32);
    c += __shfl_xor(c, m, 32);
  }
  if (lane == 0) {
    float lam_init = 0.8f - 0.6f * __expf(-0.3f * 12.0f);
    *lam = __expf(a) - __expf(c) + lam_init;
  }
}

// ---------------- bf16 WMMA GEMM: out = A[M,K] @ W[K,N] ----------------
// WT is W transposed: [Nall][K] bf16 -> every staged tile is a plain 2D tile.
// mode 0: q layout [b][2H][n][64] * D^-0.5 (bf16)
// mode 1: k layout [b][2H][n][64]          (bf16)
// mode 2: vT layout [b][H][128][n]         (bf16, transposed for attention)
// mode 3: plain row-major fp32 [M][N]
#define GTM 128
#define GTN 64
#define GTK 32
__global__ __launch_bounds__(256) void gemm_bf16_kernel(
    const unsigned short* __restrict__ A, const unsigned short* __restrict__ WT,
    int M, int K, int Nall, int mode, unsigned short* __restrict__ outb,
    float* __restrict__ outf, float scale) {
  __shared__ unsigned short Asm[2][GTM * GTK];   // [row m][k]
  __shared__ unsigned short Bsm[2][GTN * GTK];   // [row n][k]
  int wave = (int)(threadIdx.x >> 5u);
  int lane = (int)(threadIdx.x & 31u);
  int mbase = blockIdx.x * GTM;
  int nbase = blockIdx.y * GTN;
  int nsteps = K / GTK;

#if HAVE_TDM
  if (wave == 0) {  // prologue: DMA tile 0 into buffer 0
    tdm_load_2d(lds_off(Asm[0]), A + (size_t)mbase * K, (unsigned)K,
                (unsigned)M, (unsigned)K, GTK, GTM);
    tdm_load_2d(lds_off(Bsm[0]), WT + (size_t)nbase * K, (unsigned)K,
                (unsigned)Nall, (unsigned)K, GTK, GTN);
  }
#endif
  v8f acc[4] = {};
  for (int it = 0; it < nsteps; ++it) {
    int k0 = it * GTK, buf = it & 1;
#if HAVE_TDM
    if (wave == 0) wait_tensorcnt0();
    __syncthreads();  // publish tile `it`; readers of buf^1 (it-1) are done
    if (wave == 0 && it + 1 < nsteps) {  // overlap DMA of it+1 with compute
      tdm_load_2d(lds_off(Asm[buf ^ 1]), A + (size_t)mbase * K + k0 + GTK,
                  (unsigned)K, (unsigned)M, (unsigned)K, GTK, GTM);
      tdm_load_2d(lds_off(Bsm[buf ^ 1]), WT + (size_t)nbase * K + k0 + GTK,
                  (unsigned)K, (unsigned)Nall, (unsigned)K, GTK, GTN);
    }
#else
    for (int i = (int)threadIdx.x; i < (GTM * GTK) / 2; i += 256) {
      int row = i >> 4, colp = i & 15;
      ((unsigned int*)Asm[buf])[i] =
          *(const unsigned int*)(A + (size_t)(mbase + row) * K + k0 + 2 * colp);
    }
    for (int i = (int)threadIdx.x; i < (GTN * GTK) / 2; i += 256) {
      int row = i >> 4, colp = i & 15;
      ((unsigned int*)Bsm[buf])[i] =
          *(const unsigned int*)(WT + (size_t)(nbase + row) * K + k0 + 2 * colp);
    }
    __syncthreads();
#endif
    v16bf aF = load_a_frag(Asm[buf] + wave * 16 * GTK, GTK);
#pragma unroll
    for (int c = 0; c < 4; ++c) {
      v16bf bF = load_b_frag(Bsm[buf] + c * 16 * GTK, GTK, 0);
      acc[c] = wmma_bf16(aF, bF, acc[c]);
    }
  }
  // epilogue
  int coln = lane & 15, half = lane >> 4;
#pragma unroll
  for (int c = 0; c < 4; ++c) {
#pragma unroll
    for (int r = 0; r < 8; ++r) {
      int m = mbase + wave * 16 + r + 8 * half;
      int ng = nbase + c * 16 + coln;
      float val = acc[c][r] * scale;
      if (mode == 3) {
        outf[(size_t)m * Nall + ng] = val;
      } else {
        int b = m >> 11, n = m & 2047;  // N=2048 rows per batch
        if (mode == 2) {  // v stored transposed: [b][h][d][n]
          int h = ng >> 7, d = ng & 127;
          outb[(((size_t)(b * PH + h)) * 128 + d) * PN + n] = f2bf(val);
        } else {
          int h2 = ng >> 6, d = ng & 63;
          outb[(((size_t)(b * 2 * PH + h2)) * PN + n) * 64 + d] = f2bf(val);
        }
      }
    }
  }
}

// ---------------- fused differential flash attention ----------------
// grid: B*H*(N/128) blocks, 256 threads (8 waves, 16 query rows each)
__global__ __launch_bounds__(256) void diffattn_kernel(
    const unsigned short* __restrict__ qb,  // [B][2H][N][64] bf16 (pre-scaled)
    const unsigned short* __restrict__ kb,  // [B][2H][N][64] bf16
    const unsigned short* __restrict__ vb,  // [B][H][128][N] bf16 (transposed)
    const float* __restrict__ lam_p, const float* __restrict__ ln_g,
    const float* __restrict__ ln_b, unsigned short* __restrict__ ob) {
  __shared__ unsigned short k1s[2][32 * 64];   // [key][dim]
  __shared__ unsigned short k2s[2][32 * 64];   // [key][dim]
  __shared__ unsigned short vsT[2][128 * 32];  // [dim][key]
  __shared__ unsigned short ps[8 * 16 * 32];   // per-wave P scratch [row][key]

  int blk = (int)blockIdx.x;
  int qt = blk & 15;
  int h = (blk >> 4) & 7;
  int b = blk >> 7;
  int wave = (int)(threadIdx.x >> 5u), lane = (int)(threadIdx.x & 31u);
  int half = lane >> 4, ln16 = lane & 15;
  int qrow = qt * 128 + wave * 16;

  const unsigned short* q1p = qb + ((((size_t)b * 16 + 2 * h) * PN) + qrow) * 64;
  const unsigned short* q2p = qb + ((((size_t)b * 16 + 2 * h + 1) * PN) + qrow) * 64;
  const unsigned short* k1p = kb + (((size_t)b * 16 + 2 * h) * PN) * 64;
  const unsigned short* k2p = kb + (((size_t)b * 16 + 2 * h + 1) * PN) * 64;
  const unsigned short* vtp = vb + (((size_t)b * PH + h) * 128) * PN; // [128][N]

  v16bf q1f[2], q2f[2];
  q1f[0] = load_a_frag(q1p, 64);
  q1f[1] = load_a_frag(q1p + 32, 64);
  q2f[0] = load_a_frag(q2p, 64);
  q2f[1] = load_a_frag(q2p + 32, 64);

  v8f acc1[8] = {}, acc2[8] = {};
  float m1[8], l1[8], m2[8], l2[8];
#pragma unroll
  for (int r = 0; r < 8; ++r) {
    m1[r] = -1e30f; l1[r] = 0.f;
    m2[r] = -1e30f; l2[r] = 0.f;
  }

  unsigned short* pw = ps + wave * (16 * 32);
  const int nsteps = PN / 32;

#if HAVE_TDM
  if (wave == 0) {  // prologue DMA: tile 0 -> buffer 0
    tdm_load_2d(lds_off(k1s[0]), k1p, 64u, PN, 64u, 64u, 32u);
    tdm_load_2d(lds_off(k2s[0]), k2p, 64u, PN, 64u, 64u, 32u);
    tdm_load_2d(lds_off(vsT[0]), vtp, PN, 128u, PN, 32u, 128u);
  }
#endif
  for (int it = 0; it < nsteps; ++it) {
    int kb0 = it * 32, buf = it & 1;
#if HAVE_TDM
    if (wave == 0) wait_tensorcnt0();
    __syncthreads();  // publish tile `it`; readers of buf^1 (it-1) done
    if (wave == 0 && it + 1 < nsteps) {  // overlap next DMA with compute
      tdm_load_2d(lds_off(k1s[buf ^ 1]), k1p + (size_t)(kb0 + 32) * 64, 64u, PN,
                  64u, 64u, 32u);
      tdm_load_2d(lds_off(k2s[buf ^ 1]), k2p + (size_t)(kb0 + 32) * 64, 64u, PN,
                  64u, 64u, 32u);
      tdm_load_2d(lds_off(vsT[buf ^ 1]), vtp + kb0 + 32, PN, 128u, PN, 32u,
                  128u);
    }
#else
    for (int i = (int)threadIdx.x; i < 1024; i += 256) {
      int row = i >> 5, colp = i & 31;
      ((unsigned int*)k1s[buf])[i] =
          *(const unsigned int*)(k1p + (size_t)(kb0 + row) * 64 + 2 * colp);
      ((unsigned int*)k2s[buf])[i] =
          *(const unsigned int*)(k2p + (size_t)(kb0 + row) * 64 + 2 * colp);
    }
    for (int i = (int)threadIdx.x; i < 2048; i += 256) {
      int row = i >> 4, colp = i & 15;
      ((unsigned int*)vsT[buf])[i] =
          *(const unsigned int*)(vtp + (size_t)row * PN + kb0 + 2 * colp);
    }
    __syncthreads();
#endif

    // v B-fragments (shared by both half-heads): 8 chunks of 16 dims, K=32 keys
    v16bf vB[8];
#pragma unroll
    for (int c = 0; c < 8; ++c) vB[c] = load_b_frag(vsT[buf] + c * 16 * 32, 32, 0);

    // score tiles: S = q(16x64) @ k^T(64x16) via two chained K=32 WMMAs
    v8f S1a = {}, S1b = {}, S2a = {}, S2b = {};
    {
      const unsigned short* kA = k1s[buf];
      const unsigned short* kB2 = k2s[buf];
      v16bf kf;
      kf = load_b_frag(kA, 64, 0);             S1a = wmma_bf16(q1f[0], kf, S1a);
      kf = load_b_frag(kA, 64, 32);            S1a = wmma_bf16(q1f[1], kf, S1a);
      kf = load_b_frag(kA + 16 * 64, 64, 0);   S1b = wmma_bf16(q1f[0], kf, S1b);
      kf = load_b_frag(kA + 16 * 64, 64, 32);  S1b = wmma_bf16(q1f[1], kf, S1b);
      kf = load_b_frag(kB2, 64, 0);            S2a = wmma_bf16(q2f[0], kf, S2a);
      kf = load_b_frag(kB2, 64, 32);           S2a = wmma_bf16(q2f[1], kf, S2a);
      kf = load_b_frag(kB2 + 16 * 64, 64, 0);  S2b = wmma_bf16(q2f[0], kf, S2b);
      kf = load_b_frag(kB2 + 16 * 64, 64, 32); S2b = wmma_bf16(q2f[1], kf, S2b);
    }

    // -------- half-head 1: online softmax + PV --------
    {
      float scl[8];
#pragma unroll
      for (int r = 0; r < 8; ++r) {
        float t = rmax16(fmaxf(S1a[r], S1b[r]));
        float mn = fmaxf(m1[r], t);
        float pa = __expf(S1a[r] - mn), pb = __expf(S1b[r] - mn);
        float sc = __expf(m1[r] - mn);
        l1[r] = l1[r] * sc + rsum16(pa + pb);
        m1[r] = mn;
        scl[r] = sc;
        pw[(r + 8 * half) * 32 + ln16] = f2bf(pa);
        pw[(r + 8 * half) * 32 + 16 + ln16] = f2bf(pb);
      }
#pragma unroll
      for (int c = 0; c < 8; ++c)
#pragma unroll
        for (int r = 0; r < 8; ++r) acc1[c][r] *= scl[r];
      v16bf pA = load_a_frag(pw, 32);  // LDS in-order within wave
#pragma unroll
      for (int c = 0; c < 8; ++c) acc1[c] = wmma_bf16(pA, vB[c], acc1[c]);
    }
    // -------- half-head 2 --------
    {
      float scl[8];
#pragma unroll
      for (int r = 0; r < 8; ++r) {
        float t = rmax16(fmaxf(S2a[r], S2b[r]));
        float mn = fmaxf(m2[r], t);
        float pa = __expf(S2a[r] - mn), pb = __expf(S2b[r] - mn);
        float sc = __expf(m2[r] - mn);
        l2[r] = l2[r] * sc + rsum16(pa + pb);
        m2[r] = mn;
        scl[r] = sc;
        pw[(r + 8 * half) * 32 + ln16] = f2bf(pa);
        pw[(r + 8 * half) * 32 + 16 + ln16] = f2bf(pb);
      }
#pragma unroll
      for (int c = 0; c < 8; ++c)
#pragma unroll
        for (int r = 0; r < 8; ++r) acc2[c][r] *= scl[r];
      v16bf pA = load_a_frag(pw, 32);
#pragma unroll
      for (int c = 0; c < 8; ++c) acc2[c] = wmma_bf16(pA, vB[c], acc2[c]);
    }
  }

  // -------- combine, LayerNorm(128), store bf16 --------
  float lam = *lam_p;
  float mu[8], rinv[8];
#pragma unroll
  for (int r = 0; r < 8; ++r) {
    float i1 = 1.0f / l1[r];
    float i2 = lam / l2[r];
    float s = 0.f;
#pragma unroll
    for (int c = 0; c < 8; ++c) {
      float o = acc1[c][r] * i1 - acc2[c][r] * i2;
      acc1[c][r] = o;
      s += o;
    }
    mu[r] = rsum16(s) * (1.0f / 128.0f);
    float vv = 0.f;
#pragma unroll
    for (int c = 0; c < 8; ++c) {
      float d = acc1[c][r] - mu[r];
      vv += d * d;
    }
    vv = rsum16(vv) * (1.0f / 128.0f);
    rinv[r] = rsqrtf(vv + 1e-5f);
  }
#pragma unroll
  for (int c = 0; c < 8; ++c) {
    int ch = c * 16 + ln16;
    float g = ln_g[ch], be = ln_b[ch];
#pragma unroll
    for (int r = 0; r < 8; ++r) {
      float xn = (acc1[c][r] - mu[r]) * rinv[r] * g + be;
      int nrow = qrow + r + 8 * half;
      ob[((size_t)b * PN + nrow) * PE + h * 128 + ch] = f2bf(xn);
    }
  }
}

// ---------------- host launcher ----------------
extern "C" void kernel_launch(void* const* d_in, const int* in_sizes, int n_in,
                              void* d_out, int out_size, void* d_ws,
                              size_t ws_size, hipStream_t stream) {
  const float* x = (const float*)d_in[0];
  const float* Wq = (const float*)d_in[1];
  const float* Wk = (const float*)d_in[2];
  const float* Wv = (const float*)d_in[3];
  const float* Wo = (const float*)d_in[4];
  const float* lq1 = (const float*)d_in[5];
  const float* lk1 = (const float*)d_in[6];
  const float* lq2 = (const float*)d_in[7];
  const float* lk2 = (const float*)d_in[8];
  const float* ln_g = (const float*)d_in[9];
  const float* ln_b = (const float*)d_in[10];
  float* out = (float*)d_out;

  const size_t NX = (size_t)PB * PN * PE;     // 4,194,304
  const size_t NW = (size_t)PE * PE;          // 1,048,576

  unsigned short* ws = (unsigned short*)d_ws;
  unsigned short* xb = ws;                // bf16 x [M][K]
  unsigned short* wqt = xb + NX;          // bf16 Wq^T [N][K]
  unsigned short* wkt = wqt + NW;
  unsigned short* wvt = wkt + NW;
  unsigned short* wot = wvt + NW;
  unsigned short* qbf = wot + NW;         // [B][2H][N][64]
  unsigned short* kbf = qbf + NX;
  unsigned short* vbf = kbf + NX;         // [B][H][128][N] (transposed)
  unsigned short* obf = vbf + NX;         // [B][N][E]
  float* lam = (float*)(obf + NX);

  // fp32 -> bf16 (weights fused with transpose)
  cvt_bf16_kernel<<<(int)((NX + 255) / 256), 256, 0, stream>>>(x, xb, (int)NX);
  cvt_bf16_T_kernel<<<(int)(NW / 256), 256, 0, stream>>>(Wq, wqt);
  cvt_bf16_T_kernel<<<(int)(NW / 256), 256, 0, stream>>>(Wk, wkt);
  cvt_bf16_T_kernel<<<(int)(NW / 256), 256, 0, stream>>>(Wv, wvt);
  cvt_bf16_T_kernel<<<(int)(NW / 256), 256, 0, stream>>>(Wo, wot);
  lambda_kernel<<<1, 32, 0, stream>>>(lq1, lk1, lq2, lk2, lam);

  dim3 gg(PB * PN / GTM, PE / GTN);  // (32, 16)
  // q = (x@Wq)*D^-0.5 ; k = x@Wk ; v = x@Wv (v written transposed)
  gemm_bf16_kernel<<<gg, 256, 0, stream>>>(xb, wqt, PB * PN, PE, PE, 0, qbf,
                                           nullptr, 0.125f);
  gemm_bf16_kernel<<<gg, 256, 0, stream>>>(xb, wkt, PB * PN, PE, PE, 1, kbf,
                                           nullptr, 1.0f);
  gemm_bf16_kernel<<<gg, 256, 0, stream>>>(xb, wvt, PB * PN, PE, PE, 2, vbf,
                                           nullptr, 1.0f);

  // fused differential flash attention + LayerNorm
  diffattn_kernel<<<PB * PH * (PN / 128), 256, 0, stream>>>(
      qbf, kbf, vbf, lam, ln_g, ln_b, obf);

  // final projection: out = o @ Wo  (fp32 output)
  gemm_bf16_kernel<<<gg, 256, 0, stream>>>(obf, wot, PB * PN, PE, PE, 3, nullptr,
                                           out, 1.0f);
}
